// Decoder_5188320494316
// MI455X (gfx1250) — matrix operand order repaired and checked by reading, not compiled, not run on hardware
//
#include <hip/hip_runtime.h>
#include <hip/hip_bf16.h>
#include <math.h>

typedef __attribute__((ext_vector_type(16))) __bf16 v16bf;
typedef __attribute__((ext_vector_type(8)))  __bf16 v8bf;
typedef __attribute__((ext_vector_type(8)))  float  v8f;

#define BATCH   8192
#define UNITS   1024
#define GATES   3072
#define ANGLE   12
#define RADIUS  64
#define NPTS    30

// d_out section offsets (floats): a_oh, a_prob, r_oh, r_prob
#define A_N   ((size_t)BATCH * NPTS * ANGLE)    // 2,949,120
#define R_N   ((size_t)BATCH * NPTS * RADIUS)   // 15,728,640
#define OFF_AOH   ((size_t)0)
#define OFF_APROB (A_N)
#define OFF_ROH   (2 * A_N)
#define OFF_RPROB (2 * A_N + R_N)

#define WMMA_BF16(A, B, C) \
  __builtin_amdgcn_wmma_f32_16x16x32_bf16(false, (A), false, (B), (short)0, (C), false, false)

// ---------------------------------------------------------------------------
// deterministic Gumbel noise (hash-based; replaces jax threefry stream)
// ---------------------------------------------------------------------------
__device__ __forceinline__ float gumbel_noise(unsigned t, unsigned row,
                                              unsigned j, unsigned salt) {
  unsigned x = t * 0x9E3779B9u ^ row * 0x85EBCA6Bu ^ j * 0xC2B2AE35u ^ salt;
  x ^= x >> 16; x *= 0x7FEB352Du;
  x ^= x >> 15; x *= 0x846CA68Bu;
  x ^= x >> 16;
  float u = (float)(x >> 8) * (1.0f / 16777216.0f) + 1e-7f;  // (0,1)
  return -logf(-logf(u));
}

// A-fragment loader: lane%16 = M row, lane/16 selects K-halves {0-7,16-23} / {8-15,24-31}
__device__ __forceinline__ v16bf load_a_frag(const __bf16* __restrict__ base,
                                             int row, int kofs) {
  const __bf16* p = base + (size_t)row * UNITS + kofs;
  v8bf c0 = *(const v8bf*)(p);
  v8bf c1 = *(const v8bf*)(p + 16);
  v16bf a;
#pragma unroll
  for (int i = 0; i < 8; ++i) { a[i] = c0[i]; a[i + 8] = c1[i]; }
  return a;
}

// ---------------------------------------------------------------------------
// setup kernels
// ---------------------------------------------------------------------------
__global__ void convert_bf16_kernel(const float* __restrict__ s,
                                    __bf16* __restrict__ d, int n) {
  int i = blockIdx.x * blockDim.x + threadIdx.x;
  if (i < n) d[i] = (__bf16)s[i];
}

// W_ang [1024,12] -> bf16 [1024,16] zero padded
__global__ void convert_wang_kernel(const float* __restrict__ s,
                                    __bf16* __restrict__ d) {
  int i = blockIdx.x * blockDim.x + threadIdx.x;   // over 1024*16
  int r = i >> 4, c = i & 15;
  d[i] = (c < ANGLE) ? (__bf16)s[r * ANGLE + c] : (__bf16)0.0f;
}

__global__ void zero_bf16_kernel(__bf16* __restrict__ d, int n) {
  int i = blockIdx.x * blockDim.x + threadIdx.x;
  if (i < n) d[i] = (__bf16)0.0f;
}

// ---------------------------------------------------------------------------
// Fused GRU step:  h' = z*h + (1-z)*tanh(xWn + bn + r*(hUn)),
//   z = sig(xWz + hUz + bz), r = sig(xWr + hUr + br)
// Wave tile: 32 rows x 16 unit-cols (two M halves share the 6 B fragments).
// grid (8192/256 row-blocks, 64 col-tiles), block = 256 (8 waves)
// ---------------------------------------------------------------------------
__global__ __launch_bounds__(256)
void gru_kernel(const __bf16* __restrict__ X, const __bf16* __restrict__ H,
                const __bf16* __restrict__ Wg, const __bf16* __restrict__ Ug,
                const float* __restrict__ b_gru, __bf16* __restrict__ Hn) {
  const int lane = threadIdx.x & 31;
  const int wave = threadIdx.x >> 5;
  const int rowbase = blockIdx.x * 256 + wave * 32;
  const int col = blockIdx.y * 16;

  const int ar0 = rowbase + (lane & 15);        // M half 0
  const int ar1 = ar0 + 16;                     // M half 1
  const int akb = (lane >> 4) * 8;

  v8f az0 = {}, ar_0 = {}, axn0 = {}, ahn0 = {};
  v8f az1 = {}, ar_1 = {}, axn1 = {}, ahn1 = {};

  for (int kc = 0; kc < UNITS; kc += 32) {
    const int kofs = kc + akb;
    v16bf ax0 = load_a_frag(X, ar0, kofs);
    v16bf ax1 = load_a_frag(X, ar1, kofs);
    v16bf ah0 = load_a_frag(H, ar0, kofs);
    v16bf ah1 = load_a_frag(H, ar1, kofs);

    // B fragments: lane%16 = N col, elements = K in same order as A fragment
    v16bf bwz, bwr, bwn, buz, bur, bun;
#pragma unroll
    for (int i = 0; i < 16; ++i) {
      const size_t krow = (size_t)(kofs + (i < 8 ? i : i + 8)) * GATES + col + (lane & 15);
      bwz[i] = Wg[krow];
      bwr[i] = Wg[krow + UNITS];
      bwn[i] = Wg[krow + 2 * UNITS];
      buz[i] = Ug[krow];
      bur[i] = Ug[krow + UNITS];
      bun[i] = Ug[krow + 2 * UNITS];
    }

    az0  = WMMA_BF16(ax0, bwz, az0);
    az1  = WMMA_BF16(ax1, bwz, az1);
    az0  = WMMA_BF16(ah0, buz, az0);
    az1  = WMMA_BF16(ah1, buz, az1);
    ar_0 = WMMA_BF16(ax0, bwr, ar_0);
    ar_1 = WMMA_BF16(ax1, bwr, ar_1);
    ar_0 = WMMA_BF16(ah0, bur, ar_0);
    ar_1 = WMMA_BF16(ah1, bur, ar_1);
    axn0 = WMMA_BF16(ax0, bwn, axn0);
    axn1 = WMMA_BF16(ax1, bwn, axn1);
    ahn0 = WMMA_BF16(ah0, bun, ahn0);
    ahn1 = WMMA_BF16(ah1, bun, ahn1);
  }

  // epilogue: C/D layout -> lane%16 = N col, VGPR g + 8*(lane/16) = M row
  const int n  = lane & 15;
  const int cz = col + n;
  const float bz = b_gru[cz];
  const float br = b_gru[UNITS + cz];
  const float bn = b_gru[2 * UNITS + cz];
  const int moff = (lane >> 4) << 3;

#pragma unroll
  for (int half = 0; half < 2; ++half) {
    const v8f& az  = half ? az1  : az0;
    const v8f& arr = half ? ar_1 : ar_0;
    const v8f& axn = half ? axn1 : axn0;
    const v8f& ahn = half ? ahn1 : ahn0;
#pragma unroll
    for (int g = 0; g < 8; ++g) {
      const int row = rowbase + half * 16 + g + moff;
      const float hold = (float)H[(size_t)row * UNITS + cz];
      const float z = 1.0f / (1.0f + __expf(-(az[g] + bz)));
      const float r = 1.0f / (1.0f + __expf(-(arr[g] + br)));
      const float nn = tanhf(axn[g] + bn + r * ahn[g]);
      Hn[(size_t)row * UNITS + cz] = (__bf16)(z * hold + (1.0f - z) * nn);
    }
  }
}

// ---------------------------------------------------------------------------
// Head: logits (WMMA) -> softmax -> Gumbel-argmax sample -> one-hot & probs
// grid 256 blocks x 64 threads (2 waves), one 16-row tile per wave
// ---------------------------------------------------------------------------
__global__ __launch_bounds__(64)
void head_kernel(const __bf16* __restrict__ H,
                 const __bf16* __restrict__ Wang, const __bf16* __restrict__ Wrad,
                 const float* __restrict__ b_ang, const float* __restrict__ b_rad,
                 float* __restrict__ out, int t,
                 int* __restrict__ a_idx, int* __restrict__ r_idx) {
  __shared__ float lds[2][16][80];   // per wave: 16 rows x (16 angle-pad + 64 radius)
  const int lane = threadIdx.x & 31;
  const int wave = threadIdx.x >> 5;
  const int rowbase = (blockIdx.x * 2 + wave) * 16;
  const int ar  = rowbase + (lane & 15);
  const int akb = (lane >> 4) * 8;

  v8f aa = {}, r0 = {}, r1 = {}, r2 = {}, r3 = {};

  for (int kc = 0; kc < UNITS; kc += 32) {
    v16bf ah = load_a_frag(H, ar, kc + akb);

    v16bf ba, b0, b1, b2, b3;
#pragma unroll
    for (int i = 0; i < 16; ++i) {
      const int k = kc + akb + (i < 8 ? i : i + 8);
      ba[i] = Wang[(size_t)k * 16 + (lane & 15)];
      b0[i] = Wrad[(size_t)k * RADIUS + (lane & 15)];
      b1[i] = Wrad[(size_t)k * RADIUS + 16 + (lane & 15)];
      b2[i] = Wrad[(size_t)k * RADIUS + 32 + (lane & 15)];
      b3[i] = Wrad[(size_t)k * RADIUS + 48 + (lane & 15)];
    }

    aa = WMMA_BF16(ah, ba, aa);
    r0 = WMMA_BF16(ah, b0, r0);
    r1 = WMMA_BF16(ah, b1, r1);
    r2 = WMMA_BF16(ah, b2, r2);
    r3 = WMMA_BF16(ah, b3, r3);
  }

  const int n = lane & 15;
  const int moff = (lane >> 4) << 3;
#pragma unroll
  for (int g = 0; g < 8; ++g) {
    const int m = g + moff;
    lds[wave][m][n]      = aa[g];
    lds[wave][m][16 + n] = r0[g];
    lds[wave][m][32 + n] = r1[g];
    lds[wave][m][48 + n] = r2[g];
    lds[wave][m][64 + n] = r3[g];
  }
  __syncthreads();

  if (lane < 16) {
    const int row = rowbase + lane;
    const size_t rb_a = ((size_t)row * NPTS + t) * ANGLE;
    const size_t rb_r = ((size_t)row * NPTS + t) * RADIUS;

    // ---- angle head (12) ----
    {
      float mx = -3.4e38f;
      for (int j = 0; j < ANGLE; ++j) {
        float l = lds[wave][lane][j] + b_ang[j];
        mx = fmaxf(mx, l);
      }
      float s = 0.0f; int best = 0; float bg = -3.4e38f;
      for (int j = 0; j < ANGLE; ++j) {
        float l = lds[wave][lane][j] + b_ang[j];
        float e = __expf(l - mx);
        s += e;
        lds[wave][lane][j] = e;
        float gv = l + gumbel_noise((unsigned)t, (unsigned)row, (unsigned)j, 0xA17Eu);
        if (gv > bg) { bg = gv; best = j; }
      }
      const float inv = 1.0f / s;
      for (int j = 0; j < ANGLE; ++j) {
        out[OFF_APROB + rb_a + j] = lds[wave][lane][j] * inv;
        out[OFF_AOH   + rb_a + j] = (j == best) ? 1.0f : 0.0f;
      }
      a_idx[row] = best;
    }
    // ---- radius head (64) ----
    {
      float mx = -3.4e38f;
      for (int j = 0; j < RADIUS; ++j) {
        float l = lds[wave][lane][16 + j] + b_rad[j];
        mx = fmaxf(mx, l);
      }
      float s = 0.0f; int best = 0; float bg = -3.4e38f;
      for (int j = 0; j < RADIUS; ++j) {
        float l = lds[wave][lane][16 + j] + b_rad[j];
        float e = __expf(l - mx);
        s += e;
        lds[wave][lane][16 + j] = e;
        float gv = l + gumbel_noise((unsigned)t, (unsigned)row, (unsigned)j, 0x5ADu);
        if (gv > bg) { bg = gv; best = j; }
      }
      const float inv = 1.0f / s;
      for (int j = 0; j < RADIUS; ++j) {
        out[OFF_RPROB + rb_r + j] = lds[wave][lane][16 + j] * inv;
        out[OFF_ROH   + rb_r + j] = (j == best) ? 1.0f : 0.0f;
      }
      r_idx[row] = best;
    }
  }
}

// ---------------------------------------------------------------------------
// x_next = W_enc[a_idx] + W_enc[12 + r_idx] + b_enc   (one-hot @ W_enc == gather)
// ---------------------------------------------------------------------------
__global__ __launch_bounds__(256)
void build_x_kernel(const int* __restrict__ a_idx, const int* __restrict__ r_idx,
                    const float* __restrict__ W_enc, const float* __restrict__ b_enc,
                    __bf16* __restrict__ X) {
  int i = blockIdx.x * blockDim.x + threadIdx.x;    // over BATCH*UNITS
  int row = i >> 10;
  int c = i & (UNITS - 1);
  float v = W_enc[(size_t)a_idx[row] * UNITS + c] +
            W_enc[(size_t)(ANGLE + r_idx[row]) * UNITS + c] + b_enc[c];
  X[i] = (__bf16)v;
}

// ---------------------------------------------------------------------------
extern "C" void kernel_launch(void* const* d_in, const int* in_sizes, int n_in,
                              void* d_out, int out_size, void* d_ws, size_t ws_size,
                              hipStream_t stream) {
  (void)in_sizes; (void)n_in; (void)out_size; (void)ws_size;

  const float* s     = (const float*)d_in[0];
  const float* W_enc = (const float*)d_in[1];
  const float* b_enc = (const float*)d_in[2];
  const float* W_gru = (const float*)d_in[3];
  const float* U_gru = (const float*)d_in[4];
  const float* b_gru = (const float*)d_in[5];
  const float* W_ang = (const float*)d_in[6];
  const float* b_ang = (const float*)d_in[7];
  const float* W_rad = (const float*)d_in[8];
  const float* b_rad = (const float*)d_in[9];
  float* out = (float*)d_out;

  // workspace carving (256B aligned)
  char* p = (char*)d_ws;
  size_t off = 0;
  auto carve = [&](size_t bytes) {
    void* r = p + off;
    off = (off + bytes + 255) & ~(size_t)255;
    return r;
  };
  __bf16* Wg   = (__bf16*)carve((size_t)UNITS * GATES * 2);   // 6 MB
  __bf16* Ug   = (__bf16*)carve((size_t)UNITS * GATES * 2);   // 6 MB
  __bf16* Wang = (__bf16*)carve((size_t)UNITS * 16 * 2);
  __bf16* Wrad = (__bf16*)carve((size_t)UNITS * RADIUS * 2);
  int*    aidx = (int*)carve((size_t)BATCH * 4);
  int*    ridx = (int*)carve((size_t)BATCH * 4);
  __bf16* X    = (__bf16*)carve((size_t)BATCH * UNITS * 2);   // 16 MB
  __bf16* H0   = (__bf16*)carve((size_t)BATCH * UNITS * 2);   // 16 MB
  __bf16* H1   = (__bf16*)carve((size_t)BATCH * UNITS * 2);   // 16 MB

  // setup: convert weights / initial state to bf16
  {
    int n = UNITS * GATES;
    convert_bf16_kernel<<<n / 256, 256, 0, stream>>>(W_gru, Wg, n);
    convert_bf16_kernel<<<n / 256, 256, 0, stream>>>(U_gru, Ug, n);
    convert_wang_kernel<<<(UNITS * 16) / 256, 256, 0, stream>>>(W_ang, Wang);
    n = UNITS * RADIUS;
    convert_bf16_kernel<<<n / 256, 256, 0, stream>>>(W_rad, Wrad, n);
    n = BATCH * UNITS;
    convert_bf16_kernel<<<n / 256, 256, 0, stream>>>(s, H0, n);
    zero_bf16_kernel<<<n / 256, 256, 0, stream>>>(X, n);
  }

  __bf16* hcur = H0;
  __bf16* hnext = H1;
  for (int t = 0; t < NPTS; ++t) {
    gru_kernel<<<dim3(BATCH / 256, UNITS / 16), 256, 0, stream>>>(
        X, hcur, Wg, Ug, b_gru, hnext);
    head_kernel<<<BATCH / 32, 64, 0, stream>>>(
        hnext, Wang, Wrad, b_ang, b_rad, out, t, aidx, ridx);
    build_x_kernel<<<(BATCH * UNITS) / 256, 256, 0, stream>>>(
        aidx, ridx, W_enc, b_enc, X);
    __bf16* tmp = hcur; hcur = hnext; hnext = tmp;
  }
}
